// Mutual_ODEncoder_45681272160421
// MI455X (gfx1250) — matrix operand order repaired and checked by reading, not compiled, not run on hardware
//
#include <hip/hip_runtime.h>
#include <hip/hip_bf16.h>
#include <math.h>

// ---------------------------------------------------------------------------
// Dims (fixed by the reference)
//   B=16, C=COUT=64, H=W=128, K=3, KN=4, A=16, T=34, pad=1
// Implicit GEMM per (branch, batch): M=64, N=16384, K=576 (ordered chunk*288+p*32+cin%32, chunk=cin/32)
// ch[b,cin] is folded into the aggregated weights (x*ch (*) w == x (*) (w*ch)).
// ---------------------------------------------------------------------------

typedef __attribute__((ext_vector_type(16))) __bf16 v16bf;
typedef __attribute__((ext_vector_type(8)))  float  v8f;

// Workspace layout (float offsets)
//   [0,1024)      mean_over  (b*64+c)
//   [1024,2048)   mean_under
//   [2048,4096)   ch  [branch][b*64+c]
//   [4096,6144)   fl  [branch][b*64+c]
//   [6144,6432)   sp  [branch][b*9+p]
//   [6432,6560)   ka  [branch][b*4+kn]
//   byte 32768+   wfrag: bf16 A-fragments, ((((br*16+b)*4+mtile)*18+kstep)*32+lane)*8 dwords
#define WS_MEAN   0
#define WS_CH     2048
#define WS_FL     4096
#define WS_SP     6144
#define WS_KA     6432
#define WFRAG_BYTE_OFF 32768

__device__ __forceinline__ unsigned short bf16_bits(float f) {
    unsigned int u = __float_as_uint(f);
    u += 0x7FFFu + ((u >> 16) & 1u);      // round-to-nearest-even
    return (unsigned short)(u >> 16);
}

// ---------------------------------------------------------------------------
// 1) Global-average-pool: channel means of x_over (blocks 0..1023) and
//    x_under (blocks 1024..2047)
// ---------------------------------------------------------------------------
__global__ void __launch_bounds__(256) pool_kernel(const float* __restrict__ x_over,
                                                   const float* __restrict__ x_under,
                                                   float* __restrict__ means) {
    __shared__ float red[256];
    int bid = blockIdx.x;
    const float* src = (bid < 1024) ? x_over : x_under;
    int ch = bid & 1023;
    const float* p = src + (size_t)ch * 16384;
    float s = 0.f;
    for (int i = threadIdx.x; i < 16384; i += 256) s += p[i];
    red[threadIdx.x] = s;
    __syncthreads();
    for (int off = 128; off > 0; off >>= 1) {
        if (threadIdx.x < (unsigned)off) red[threadIdx.x] += red[threadIdx.x + off];
        __syncthreads();
    }
    if (threadIdx.x == 0) means[bid] = red[0] * (1.f / 16384.f);
}

// ---------------------------------------------------------------------------
// 2) Attention heads (one block per branch). Train-mode BN (batch stats,
//    biased variance) over B=16, ReLU, then sigmoid/softmax heads, T=34.
// ---------------------------------------------------------------------------
struct AttParams {
    const float *fc_w, *bn_g, *bn_b, *ch_w, *ch_b, *fl_w, *fl_b, *sp_w, *sp_b, *ka_w, *ka_b;
};

__global__ void __launch_bounds__(256) attention_kernel(const float* __restrict__ ws_in,
                                                        float* __restrict__ ws_out,
                                                        AttParams p0, AttParams p1) {
    int branch = blockIdx.x;                     // 0: under-branch (r1=[over,under])
    AttParams p = branch ? p1 : p0;
    const float* mfirst  = ws_in + (branch ? 1024 : 0);
    const float* msecond = ws_in + (branch ? 0 : 1024);

    __shared__ float h[16][16];
    __shared__ float sc[16], sh[16];
    int t = threadIdx.x;
    int b = t >> 4, a = t & 15;

    {   // h = pooled @ fc_w^T   (B=16 x A=16, K=128)
        const float* w = p.fc_w + a * 128;
        float s = 0.f;
        for (int i = 0; i < 64; ++i) s += mfirst[b * 64 + i] * w[i];
        for (int i = 0; i < 64; ++i) s += msecond[b * 64 + i] * w[64 + i];
        h[b][a] = s;
    }
    __syncthreads();
    if (t < 16) {   // BN over batch dim (biased var) per feature t
        float mu = 0.f;
        for (int i = 0; i < 16; ++i) mu += h[i][t];
        mu *= (1.f / 16.f);
        float var = 0.f;
        for (int i = 0; i < 16; ++i) { float d = h[i][t] - mu; var += d * d; }
        var *= (1.f / 16.f);
        float s = p.bn_g[t] * rsqrtf(var + 1e-5f);
        sc[t] = s; sh[t] = p.bn_b[t] - mu * s;
    }
    __syncthreads();
    {   float v = h[b][a] * sc[a] + sh[a]; h[b][a] = v > 0.f ? v : 0.f; }
    __syncthreads();

    const float invT = 1.f / 34.f;
    float* chw = ws_out + WS_CH + branch * 1024;
    float* flw = ws_out + WS_FL + branch * 1024;
    float* spw = ws_out + WS_SP + branch * 144;
    float* kaw = ws_out + WS_KA + branch * 64;

    for (int idx = t; idx < 1024; idx += 256) {
        int bb = idx >> 6, c = idx & 63;
        float s1 = p.ch_b[c], s2 = p.fl_b[c];
        for (int i = 0; i < 16; ++i) {
            float hv = h[bb][i];
            s1 += hv * p.ch_w[c * 16 + i];
            s2 += hv * p.fl_w[c * 16 + i];
        }
        chw[idx] = 1.f / (1.f + __expf(-s1 * invT));
        flw[idx] = 1.f / (1.f + __expf(-s2 * invT));
    }
    if (t < 144) {
        int bb = t / 9, k = t % 9;
        float s = p.sp_b[k];
        for (int i = 0; i < 16; ++i) s += h[bb][i] * p.sp_w[k * 16 + i];
        spw[t] = 1.f / (1.f + __expf(-s * invT));
    }
    if (t < 16) {   // softmax over KN=4
        float lg[4], m = -1e30f;
        for (int kn = 0; kn < 4; ++kn) {
            float s = p.ka_b[kn];
            for (int i = 0; i < 16; ++i) s += h[t][i] * p.ka_w[kn * 16 + i];
            s *= invT;
            lg[kn] = s; m = s > m ? s : m;
        }
        float den = 0.f;
        for (int kn = 0; kn < 4; ++kn) { lg[kn] = __expf(lg[kn] - m); den += lg[kn]; }
        float r = 1.f / den;
        for (int kn = 0; kn < 4; ++kn) kaw[t * 4 + kn] = lg[kn] * r;
    }
}

// ---------------------------------------------------------------------------
// 3) Aggregate weights -> bf16 packed directly in the 16x32 WMMA A-fragment
//    layout (this code's contiguous K order): lane l row M=l&15; VGPR j / half h holds
//    Klocal = 2*j + h + 16*(l>>4).  K ordering: chunk*288 + p*32 + (cin - c0),
//    step = (chunk, p) with cin base c0 = 32*chunk.
//    ch[b,cin] is folded in here so the conv kernel stages raw activations.
//    blockIdx.x = ((branch*16+b)*4+mtile)*18+kstep ; 256 thr = 32 lanes x 8 j
// ---------------------------------------------------------------------------
__global__ void __launch_bounds__(256) aggw_kernel(const float* __restrict__ shared_w,
                                                   const float* __restrict__ ws,
                                                   unsigned int* __restrict__ wfrag) {
    int bid = blockIdx.x;
    int kstep  = bid % 18;
    int mtile  = (bid / 18) & 3;
    int b      = (bid / 72) & 15;
    int branch = bid / 1152;
    int c0 = (kstep / 9) * 32;
    int p  = kstep % 9;

    const float* spv = ws + WS_SP + branch * 144 + b * 9;
    const float* kav = ws + WS_KA + branch * 64 + b * 4;
    const float* chv = ws + WS_CH + branch * 1024 + b * 64;
    float spp = spv[p];
    float ka0 = kav[0], ka1 = kav[1], ka2 = kav[2], ka3 = kav[3];

    int t = threadIdx.x;
    int lane = t >> 3, j = t & 7;
    int M = mtile * 16 + (lane & 15);
    int kgrp = lane >> 4;

    unsigned int pack = 0;
    #pragma unroll
    for (int h = 0; h < 2; ++h) {
        int kA  = 2 * j + h + 16 * kgrp;
        int cin = c0 + kA;
        // shared_weight[kn, M, cin, ky, kx], flat ((kn*64+M)*64+cin)*9+p
        size_t i0 = ((size_t)(0 * 64 + M) * 64 + cin) * 9 + p;
        size_t i1 = ((size_t)(1 * 64 + M) * 64 + cin) * 9 + p;
        size_t i2 = ((size_t)(2 * 64 + M) * 64 + cin) * 9 + p;
        size_t i3 = ((size_t)(3 * 64 + M) * 64 + cin) * 9 + p;
        float acc = ka0 * shared_w[i0] + ka1 * shared_w[i1] +
                    ka2 * shared_w[i2] + ka3 * shared_w[i3];
        acc *= spp * chv[cin];
        pack |= (unsigned int)bf16_bits(acc) << (16 * h);
    }
    wfrag[((size_t)bid * 32 + lane) * 8 + j] = pack;
}

// ---------------------------------------------------------------------------
// 4) Implicit-GEMM dynamic conv with v_wmma_f32_16x16x32_bf16.
//    grid (512, 16, 2): blockIdx.x -> (row y, 32-px x-tile); .y=batch; .z=branch
//    256 thr = 8 waves: wave = (mtile in 0..3, nsub in 0..1) -> 64 couts x 32 px.
//    LDS input tile [3 dy][36 col pad][32 cin] bf16 (pure f32->bf16 copy,
//    128-bit interior loads, zero-padded halo).  B-fragment (32x16, lane col
//    n=l&15, K = 2j+h+16*(l>>4)) is one contiguous 32B LDS read per lane.
//    18 WMMA steps (2 cin chunks x 9 taps).  Output tile is transposed
//    through the same LDS ([64 cout][36 px pad]) for coalesced b128 stores.
// ---------------------------------------------------------------------------
__global__ void __launch_bounds__(256) conv_kernel(const float* __restrict__ x_under,
                                                   const float* __restrict__ x_over,
                                                   const float* __restrict__ ws,
                                                   const unsigned int* __restrict__ wfrag,
                                                   float* __restrict__ out) {
    __shared__ __align__(32) unsigned char smraw[64 * 36 * 4];   // 9216 B, reused
    unsigned short* tile = (unsigned short*)smraw;               // [3][36][32] bf16
    float* otile = (float*)smraw;                                // [64][36] f32

    int branch = blockIdx.z;
    int b = blockIdx.y;
    int y = blockIdx.x >> 2;
    int xbase = (blockIdx.x & 3) << 5;

    const float* xin = branch ? x_over : x_under;
    const float* flv = ws + WS_FL + branch * 1024 + b * 64;

    int tid = threadIdx.x;
    int wave = tid >> 5, lane = tid & 31;
    int mtile = wave & 3, nsub = wave >> 2;
    int kgrp = lane >> 4;

    v8f acc = {};
    size_t xbb = (size_t)b << 20;                                  // b*64*16384
    size_t wfbase = ((size_t)(branch * 16 + b) * 4 + mtile) * 18;  // fragment row

    // Prefetch this wave's weight fragments (global_prefetch_b8 path).
    #pragma unroll
    for (int ks = 0; ks < 18; ++ks)
        __builtin_prefetch((const void*)(wfrag + ((wfbase + ks) * 32 + lane) * 8), 0, 3);

    for (int chunk = 0; chunk < 2; ++chunk) {
        int c0 = chunk << 5;
        __syncthreads();
        // Stage bf16 input tile [dy][col][cin_local]; cols cover x in
        // [xbase-1, xbase+33).  960 float4 spans over [xbase-4, xbase+36).
        for (int idx = tid; idx < 960; idx += 256) {
            int q  = idx % 10;
            int t2 = idx / 10;
            int dy = t2 % 3;
            int ci = t2 / 3;
            int yy = y + dy - 1;
            int gx0 = xbase - 4 + 4 * q;
            float v0 = 0.f, v1 = 0.f, v2 = 0.f, v3 = 0.f;
            if ((unsigned)yy < 128u) {
                const float* row = xin + xbb + ((size_t)(c0 + ci) << 14) + (yy << 7);
                if (gx0 >= 0 && gx0 <= 124) {
                    float4 f = *reinterpret_cast<const float4*>(row + gx0);
                    v0 = f.x; v1 = f.y; v2 = f.z; v3 = f.w;
                } else {
                    if ((unsigned)(gx0 + 0) < 128u) v0 = row[gx0 + 0];
                    if ((unsigned)(gx0 + 1) < 128u) v1 = row[gx0 + 1];
                    if ((unsigned)(gx0 + 2) < 128u) v2 = row[gx0 + 2];
                    if ((unsigned)(gx0 + 3) < 128u) v3 = row[gx0 + 3];
                }
            }
            int colb = 4 * q - 3;                 // col = gx - (xbase-1)
            unsigned short* base = tile + dy * 36 * 32 + ci;
            if ((unsigned)(colb + 0) < 34u) base[(colb + 0) * 32] = bf16_bits(v0);
            if ((unsigned)(colb + 1) < 34u) base[(colb + 1) * 32] = bf16_bits(v1);
            if ((unsigned)(colb + 2) < 34u) base[(colb + 2) * 32] = bf16_bits(v2);
            if ((unsigned)(colb + 3) < 34u) base[(colb + 3) * 32] = bf16_bits(v3);
        }
        __syncthreads();

        #pragma unroll
        for (int p = 0; p < 9; ++p) {
            int ky = p / 3, kx = p - ky * 3;
            int kstep = chunk * 9 + p;
            v16bf afr = *reinterpret_cast<const v16bf*>(
                wfrag + ((wfbase + kstep) * 32 + lane) * 8);
            int ncol = nsub * 16 + (lane & 15) + kx;
            v16bf bfr = *reinterpret_cast<const v16bf*>(
                &tile[(ky * 36 + ncol) * 32 + 16 * kgrp]);
            acc = __builtin_amdgcn_wmma_f32_16x16x32_bf16(
                false, afr, false, bfr, (short)0, acc, false, false);
        }
    }

    // Epilogue: C/D layout — VGPR v holds M = v + 8*(lane>>4), N = lane&15.
    // Transpose through LDS and emit coalesced float4 stores along x.
    __syncthreads();
    {
        int px = nsub * 16 + (lane & 15);
        #pragma unroll
        for (int v = 0; v < 8; ++v) {
            int cout = mtile * 16 + v + 8 * kgrp;
            otile[cout * 36 + px] = acc[v] * flv[cout];
        }
    }
    __syncthreads();
    {
        size_t obase = (((size_t)branch * 16 + b) * 64) << 14;
        #pragma unroll
        for (int i = tid; i < 512; i += 256) {
            int cout = i >> 3, qx = i & 7;
            float4 f = *reinterpret_cast<const float4*>(&otile[cout * 36 + qx * 4]);
            *reinterpret_cast<float4*>(
                out + obase + ((size_t)cout << 14) + (y << 7) + xbase + qx * 4) = f;
        }
    }
}

// ---------------------------------------------------------------------------
// Host launcher. Inputs flattened in setup_inputs() order:
//   [0]=x_over [1]=x_under [2..12]=under_params{fc_w,bn_g,bn_b,ch_w,ch_b,
//   fl_w,fl_b,sp_w,sp_b,ka_w,ka_b} [13..23]=over_params(same order)
//   [24]=shared_weight.  Output = out_under ++ out_over (float32).
// ---------------------------------------------------------------------------
extern "C" void kernel_launch(void* const* d_in, const int* in_sizes, int n_in,
                              void* d_out, int out_size, void* d_ws, size_t ws_size,
                              hipStream_t stream) {
    (void)in_sizes; (void)n_in; (void)out_size; (void)ws_size;

    const float* x_over  = (const float*)d_in[0];
    const float* x_under = (const float*)d_in[1];

    AttParams pu, po;
    {
        const float** u = (const float**)&pu;
        const float** o = (const float**)&po;
        for (int i = 0; i < 11; ++i) {
            u[i] = (const float*)d_in[2 + i];
            o[i] = (const float*)d_in[13 + i];
        }
    }
    const float* shared_w = (const float*)d_in[24];

    float* ws = (float*)d_ws;
    unsigned int* wfrag = (unsigned int*)((char*)d_ws + WFRAG_BYTE_OFF);
    float* out = (float*)d_out;

    pool_kernel<<<2048, 256, 0, stream>>>(x_over, x_under, ws + WS_MEAN);
    attention_kernel<<<2, 256, 0, stream>>>(ws, ws, pu, po);
    aggw_kernel<<<2304, 256, 0, stream>>>(shared_w, ws, wfrag);
    conv_kernel<<<dim3(512, 16, 2), 256, 0, stream>>>(x_under, x_over, ws, wfrag, out);
}
